// SANet_42339787604278
// MI455X (gfx1250) — compile-verified, matrix operand order checked by reading, not run on hardware
//
#include <hip/hip_runtime.h>

typedef __attribute__((ext_vector_type(16))) __bf16 v16bf;
typedef __attribute__((ext_vector_type(8)))  float  v8f;

#define BATCH 4
#define CIN   512
#define PDIM  128
#define NPIX  4096

struct alignas(16) BF8 { __bf16 b[8]; };

// ---- fragment loaders (CDNA5 WMMA 16x16x32 bf16 layouts, ISA 7.12.2) ----

// A-frag / LDS-B-frag: source is row-major [16 rows][K], rows indexed by lane%16,
// K pattern: elems 0..7 -> K = 8h..8h+7 ; elems 8..15 -> K = 16+8h..16+8h+7
__device__ __forceinline__ v16bf frag_ld_rowmajor(const __bf16* base, int ld) {
  const int lane = threadIdx.x & 31;
  const int r = lane & 15, h = lane >> 4;
  const __bf16* p = base + (size_t)r * ld + h * 8;
  BF8 lo = *(const BF8*)(p);
  BF8 hi = *(const BF8*)(p + 16);
  v16bf f;
#pragma unroll
  for (int i = 0; i < 8; ++i) { f[i] = lo.b[i]; f[i + 8] = hi.b[i]; }
  return f;
}

// B-frag: source is row-major [K rows][cols], column = lane%16 (coalesced across lanes)
__device__ __forceinline__ v16bf frag_ld_kmajor_bf16(const __bf16* base, int ld) {
  const int lane = threadIdx.x & 31;
  const int j = lane & 15, h = lane >> 4;
  v16bf f;
#pragma unroll
  for (int i = 0; i < 8; ++i) {
    f[i]     = base[(size_t)(8 * h + i) * ld + j];
    f[i + 8] = base[(size_t)(16 + 8 * h + i) * ld + j];
  }
  return f;
}

__device__ __forceinline__ v16bf frag_ld_kmajor_f32(const float* base, int ld) {
  const int lane = threadIdx.x & 31;
  const int j = lane & 15, h = lane >> 4;
  v16bf f;
#pragma unroll
  for (int i = 0; i < 8; ++i) {
    f[i]     = (__bf16)base[(size_t)(8 * h + i) * ld + j];
    f[i + 8] = (__bf16)base[(size_t)(16 + 8 * h + i) * ld + j];
  }
  return f;
}

__device__ __forceinline__ v8f wmma_bf16(v16bf a, v16bf b, v8f c) {
  return __builtin_amdgcn_wmma_f32_16x16x32_bf16(false, a, false, b, (short)0, c,
                                                 false, false);
}

// ---- kernel 1: convert the four 128x512 weight matrices to bf16 ----
__global__ void __launch_bounds__(256)
wcvt_kernel(const float* wf, const float* wg, const float* wh, const float* wk,
            __bf16* wb) {
  int idx = blockIdx.x * 256 + threadIdx.x;        // 0 .. 4*65536-1
  int proj = idx >> 16, off = idx & 65535;
  const float* src = proj == 0 ? wf : proj == 1 ? wg : proj == 2 ? wh : wk;
  wb[idx] = (__bf16)src[off];
}

// ---- kernel 2: four projection GEMMs  D[p,n] = W[p,:]·X[:,n] + bias[p] ----
// grid: b(4) x proj(4) x ntile64(64) = 1024 blocks, 8 waves each.
// wave w owns p-rows [16w,16w+16), 4 n-tiles of 16 per block.
__global__ void __launch_bounds__(256)
proj_kernel(const float* __restrict__ content, const float* __restrict__ style,
            const __bf16* __restrict__ wb,
            const float* __restrict__ bf_, const float* __restrict__ bg_,
            const float* __restrict__ bh_, const float* __restrict__ bk_,
            float* __restrict__ F0, float* __restrict__ G0,
            __bf16* __restrict__ Fb, __bf16* __restrict__ Gb,
            __bf16* __restrict__ FbT, __bf16* __restrict__ GbT,
            __bf16* __restrict__ Hv, __bf16* __restrict__ Kv) {
  const int bid  = blockIdx.x;
  const int b    = bid >> 8;
  const int proj = (bid >> 6) & 3;
  const int n0   = (bid & 63) * 64;
  const int w    = threadIdx.x >> 5;
  const int lane = threadIdx.x & 31;
  const int p0   = w * 16;

  const float* X    = (proj == 0 || proj == 2) ? content : style;
  const float* bias = proj == 0 ? bf_ : proj == 1 ? bg_ : proj == 2 ? bh_ : bk_;
  const __bf16* Wp  = wb + (size_t)proj * PDIM * CIN + (size_t)p0 * CIN;
  const float*  Xb  = X + (size_t)b * CIN * NPIX;

  v8f acc[4];
#pragma unroll
  for (int t = 0; t < 4; ++t) acc[t] = v8f{0, 0, 0, 0, 0, 0, 0, 0};

  for (int kk = 0; kk < CIN / 32; ++kk) {
    v16bf a = frag_ld_rowmajor(Wp + kk * 32, CIN);
#pragma unroll
    for (int t = 0; t < 4; ++t) {
      v16bf bb = frag_ld_kmajor_f32(Xb + (size_t)(kk * 32) * NPIX + n0 + 16 * t, NPIX);
      acc[t] = wmma_bf16(a, bb, acc[t]);
    }
  }

  const int h = lane >> 4, j = lane & 15;
  float bv[8];
#pragma unroll
  for (int r = 0; r < 8; ++r) bv[r] = bias[p0 + r + 8 * h];

#pragma unroll
  for (int t = 0; t < 4; ++t) {
#pragma unroll
    for (int r = 0; r < 8; ++r) {
      float v = acc[t][r] + bv[r];
      int p = p0 + r + 8 * h;
      int n = n0 + 16 * t + j;
      size_t idx  = (size_t)b * PDIM * NPIX + (size_t)p * NPIX + n;
      size_t idxT = (size_t)b * NPIX * PDIM + (size_t)n * PDIM + p;
      if (proj == 0)      { F0[idx] = v; Fb[idx] = (__bf16)v; FbT[idxT] = (__bf16)v; }
      else if (proj == 1) { G0[idx] = v; Gb[idx] = (__bf16)v; GbT[idxT] = (__bf16)v; }
      else if (proj == 2) { Hv[idx] = (__bf16)v; }
      else                { Kv[idx] = (__bf16)v; }
    }
  }
}

// ---- kernel 3: softmax row stats of l[i,j] = Fᵀ(i,:)·G(:,j): rowmax, 1/rowsum ----
// one wave per 16-row i-tile, stream j over 4096 in 16-col wmma tiles.
__global__ void __launch_bounds__(256)
stats_kernel(const __bf16* __restrict__ FbT, const __bf16* __restrict__ Gb,
             float* __restrict__ rmax, float* __restrict__ rinv) {
  const int gw   = blockIdx.x * 8 + (threadIdx.x >> 5);
  const int b    = gw >> 8;
  const int i0   = (gw & 255) * 16;
  const int lane = threadIdx.x & 31;
  const int h    = lane >> 4;

  const __bf16* Abase = FbT + (size_t)b * NPIX * PDIM + (size_t)i0 * PDIM;
  const __bf16* Bbase = Gb + (size_t)b * PDIM * NPIX;

  v16bf afr[4];
#pragma unroll
  for (int kk = 0; kk < 4; ++kk) afr[kk] = frag_ld_rowmajor(Abase + kk * 32, PDIM);

  float rm[8], rs[8];
#pragma unroll
  for (int r = 0; r < 8; ++r) { rm[r] = -3.0e38f; rs[r] = 0.0f; }

  for (int jt = 0; jt < NPIX / 16; ++jt) {
    v8f acc = v8f{0, 0, 0, 0, 0, 0, 0, 0};
#pragma unroll
    for (int kk = 0; kk < 4; ++kk) {
      v16bf bb = frag_ld_kmajor_bf16(Bbase + (size_t)(kk * 32) * NPIX + jt * 16, NPIX);
      acc = wmma_bf16(afr[kk], bb, acc);
    }
#pragma unroll
    for (int r = 0; r < 8; ++r) {
      float v  = acc[r];
      float tm = v;   // reduce over the 16 columns (lanes within half) per row
      tm = fmaxf(tm, __shfl_xor(tm, 1, 32));
      tm = fmaxf(tm, __shfl_xor(tm, 2, 32));
      tm = fmaxf(tm, __shfl_xor(tm, 4, 32));
      tm = fmaxf(tm, __shfl_xor(tm, 8, 32));
      float nm = fmaxf(rm[r], tm);
      float e  = __expf(v - nm);
      e += __shfl_xor(e, 1, 32);
      e += __shfl_xor(e, 2, 32);
      e += __shfl_xor(e, 4, 32);
      e += __shfl_xor(e, 8, 32);
      rs[r] = rs[r] * __expf(rm[r] - nm) + e;
      rm[r] = nm;
    }
  }
  if ((lane & 15) == 0) {
#pragma unroll
    for (int r = 0; r < 8; ++r) {
      int i = i0 + r + 8 * h;
      rmax[b * NPIX + i] = rm[r];
      rinv[b * NPIX + i] = 1.0f / rs[r];
    }
  }
}

// ---- kernels 4/5: streaming attention aggregation (logits recomputed) ----
// Block owns (b, 16-wide output column tile m0). Wave w owns c-rows [16w,16w+16)
// AND produces the P tile for n-chunk w of each 256-wide n iteration.
// STATS_BY_COL=1: Landmark (P = exp(l - rmax[m])·rinv[m], l via GbT/Fb, val=Kv, +F0)
// STATS_BY_COL=0: Face     (P = exp(l - rmax[n])·rinv[n], l via FbT/Gb, val=Hv, +G0)
template <int STATS_BY_COL>
__global__ void __launch_bounds__(256)
attn_pass_kernel(const __bf16* __restrict__ LA,   // [B][N][128]  logits A rows (k-dim idx)
                 const __bf16* __restrict__ LB,   // [B][128][N]  logits B cols
                 const __bf16* __restrict__ VAL,  // [B][128][N]  aggregation values
                 const float* __restrict__ resid, // [B][128][N]  residual (f32)
                 const float* __restrict__ rmax, const float* __restrict__ rinv,
                 float* __restrict__ out) {
  __shared__ alignas(16) __bf16 Pt[8][16][32];   // [chunk][j=m][k=n], 8 KB

  const int b    = blockIdx.x >> 8;
  const int m0   = (blockIdx.x & 255) * 16;
  const int w    = threadIdx.x >> 5;
  const int lane = threadIdx.x & 31;
  const int h = lane >> 4, j = lane & 15;
  const int c0 = w * 16;

  const __bf16* LAb = LA + (size_t)b * NPIX * PDIM;
  const __bf16* LBb = LB + (size_t)b * PDIM * NPIX + m0;
  const __bf16* Vb  = VAL + (size_t)b * PDIM * NPIX + (size_t)c0 * NPIX;
  const float* rmb  = rmax + (size_t)b * NPIX;
  const float* rib  = rinv + (size_t)b * NPIX;

  v16bf lb[4];                       // logits B-frags: constant over all n
#pragma unroll
  for (int kk = 0; kk < 4; ++kk)
    lb[kk] = frag_ld_kmajor_bf16(LBb + (size_t)(kk * 32) * NPIX, NPIX);

  float colMax = 0.f, colInv = 0.f;
  if (STATS_BY_COL) { colMax = rmb[m0 + j]; colInv = rib[m0 + j]; }

  v8f acc = v8f{0, 0, 0, 0, 0, 0, 0, 0};

  for (int iter = 0; iter < NPIX / 256; ++iter) {
    const int nbase = iter * 256 + w * 32;
#pragma unroll
    for (int t = 0; t < 2; ++t) {
      v8f la = v8f{0, 0, 0, 0, 0, 0, 0, 0};
      const __bf16* arow = LAb + (size_t)(nbase + 16 * t) * PDIM;
#pragma unroll
      for (int kk = 0; kk < 4; ++kk) {
        v16bf a = frag_ld_rowmajor(arow + kk * 32, PDIM);
        la = wmma_bf16(a, lb[kk], la);
      }
      BF8 pk;
#pragma unroll
      for (int r = 0; r < 8; ++r) {
        float pval;
        if (STATS_BY_COL) {
          pval = __expf(la[r] - colMax) * colInv;
        } else {
          int n = nbase + 16 * t + r + 8 * h;
          pval = __expf(la[r] - rmb[n]) * rib[n];
        }
        pk.b[r] = (__bf16)pval;
      }
      // tile element (row n = 16t+8h+r, col m = j) -> Pt[w][j][16t+8h+r]
      *(BF8*)&Pt[w][j][16 * t + 8 * h] = pk;
    }
    __syncthreads();
#pragma unroll
    for (int ch = 0; ch < 8; ++ch) {
      v16bf a  = frag_ld_rowmajor(Vb + iter * 256 + ch * 32, NPIX);
      v16bf bb = frag_ld_rowmajor(&Pt[ch][0][0], 32);   // [j][k] layout
      acc = wmma_bf16(a, bb, acc);
    }
    __syncthreads();
  }

#pragma unroll
  for (int r = 0; r < 8; ++r) {
    int c = c0 + r + 8 * h;
    size_t idx = (size_t)b * PDIM * NPIX + (size_t)c * NPIX + (m0 + j);
    out[idx] = acc[r] + resid[idx];
  }
}

// ---- host launcher ----
extern "C" void kernel_launch(void* const* d_in, const int* in_sizes, int n_in,
                              void* d_out, int out_size, void* d_ws, size_t ws_size,
                              hipStream_t stream) {
  const float* content = (const float*)d_in[0];
  const float* style   = (const float*)d_in[1];
  const float* Wf = (const float*)d_in[2];
  const float* bf_ = (const float*)d_in[3];
  const float* Wg = (const float*)d_in[4];
  const float* bg_ = (const float*)d_in[5];
  const float* Wh = (const float*)d_in[6];
  const float* bh_ = (const float*)d_in[7];
  const float* Wk = (const float*)d_in[8];
  const float* bk_ = (const float*)d_in[9];
  float* out = (float*)d_out;

  const size_t PN = (size_t)BATCH * PDIM * NPIX;   // 2,097,152 elems
  char* ws = (char*)d_ws;
  size_t off = 0;
  auto carve = [&](size_t bytes) {
    void* p = ws + off;
    off += (bytes + 255) & ~(size_t)255;
    return p;
  };
  __bf16* Wb   = (__bf16*)carve((size_t)4 * PDIM * CIN * 2);
  float*  F0   = (float*)carve(PN * 4);
  float*  G0   = (float*)carve(PN * 4);
  __bf16* Fb   = (__bf16*)carve(PN * 2);
  __bf16* Gb   = (__bf16*)carve(PN * 2);
  __bf16* FbT  = (__bf16*)carve(PN * 2);
  __bf16* GbT  = (__bf16*)carve(PN * 2);
  __bf16* Hv   = (__bf16*)carve(PN * 2);
  __bf16* Kv   = (__bf16*)carve(PN * 2);
  float*  rmaxA = (float*)carve((size_t)BATCH * NPIX * 4);
  float*  rinvA = (float*)carve((size_t)BATCH * NPIX * 4);
  (void)ws_size; (void)in_sizes; (void)n_in; (void)out_size;

  wcvt_kernel<<<dim3(1024), dim3(256), 0, stream>>>(Wf, Wg, Wh, Wk, Wb);
  proj_kernel<<<dim3(1024), dim3(256), 0, stream>>>(content, style, Wb,
                                                    bf_, bg_, bh_, bk_,
                                                    F0, G0, Fb, Gb, FbT, GbT, Hv, Kv);
  stats_kernel<<<dim3(128), dim3(256), 0, stream>>>(FbT, Gb, rmaxA, rinvA);
  // Landmark_en = Kv·Sᵀ + F0  (output slot 0)
  attn_pass_kernel<1><<<dim3(1024), dim3(256), 0, stream>>>(GbT, Fb, Kv, F0,
                                                            rmaxA, rinvA, out);
  // Face_en = Hv·S + G0       (output slot 1)
  attn_pass_kernel<0><<<dim3(1024), dim3(256), 0, stream>>>(FbT, Gb, Hv, G0,
                                                            rmaxA, rinvA, out + PN);
}
